// Full32BitNibbleFFN_62380105007506
// MI455X (gfx1250) — compile-verified, matrix operand order checked by reading, not run on hardware
//
#include <hip/hip_runtime.h>

typedef __attribute__((ext_vector_type(16))) _Float16     v16h;
typedef __attribute__((ext_vector_type(8)))  float        v8f;
typedef __attribute__((ext_vector_type(2)))  float        v2f;
typedef __attribute__((ext_vector_type(8)))  unsigned int v8u;

#define NSTEP 8
#define WAVES_PER_BLOCK 2
#define BLOCK_THREADS (WAVES_PER_BLOCK * 32)
// exp(100*(x-3)) == exp2(x*SCALE_L2E + BIAS_L2E)
#define SCALE_L2E 144.26950408889634f
#define BIAS_L2E  (-432.808512266689f)

// ---- LDS layout (u32 words) -------------------------------------------------
// BIN0  : [32 tiles][32 lanes][8] u32 (f16 pairs, W_in rows 0..31)    8192 w
// BSUM  : [16 chunks][32 lanes][8] u32 (W_sum fragments)              4096 w
// BCOUT : [16 chunks][32 lanes][8] u32 (W_cout frags; col2 = ones,
//          cols>=3 = 0 -> accC col2 yields softmax denominator Z)     4096 w
// BCARRY: [32 tiles][32 lanes][2] f32 (W_in rows 32..33; half1 = 0)   2048 w
// E     : [waves][16 rows][512 cols] f16 (exp(logits) transpose buf)  8192 w
// CARRY : [waves][16 rows][2] f32                                       64 w
#define OFF_BIN0   0
#define OFF_BSUM   (OFF_BIN0 + 32*32*8)
#define OFF_BCOUT  (OFF_BSUM + 16*32*8)
#define OFF_BCARRY (OFF_BCOUT + 16*32*8)
#define OFF_E      (OFF_BCARRY + 32*32*2)
#define OFF_CARRY  (OFF_E + (WAVES_PER_BLOCK*16*512*2)/4)
#define SMEM_U32   (OFF_CARRY + WAVES_PER_BLOCK*16*2)
#define SMEM_BYTES (SMEM_U32 * 4)

__device__ __forceinline__ unsigned pack2h(float lo, float hi) {
    union { _Float16 h[2]; unsigned u; } u;
    u.h[0] = (_Float16)lo; u.h[1] = (_Float16)hi;
    return u.u;
}

// Raw v_exp_f32: skip libm's sub-normal-range fixup (cmp/cndmask/ldexp chain).
// Softmax numerators in that range are exactly the ones that must flush to 0.
__device__ __forceinline__ float fast_exp2(float x) {
#if __has_builtin(__builtin_amdgcn_exp2f)
    return __builtin_amdgcn_exp2f(x);
#else
    return exp2f(x);
#endif
}

union Frag16 { v8u u; uint4 q[2]; v16h h; };

__global__ __launch_bounds__(BLOCK_THREADS)
void nibble32_wmma_kernel(const float* __restrict__ a_oh,
                          const float* __restrict__ b_oh,
                          const float* __restrict__ Win,    // [34][512]
                          const float* __restrict__ Wsum,   // [512][16]
                          const float* __restrict__ Wcout,  // [512][2]
                          float* __restrict__ out)          // [N][8][16]
{
    extern __shared__ v8u smem_raw[];                 // 32B-aligned dynamic LDS
    unsigned*  SM     = (unsigned*)smem_raw;
    unsigned*  BIN0   = SM + OFF_BIN0;
    unsigned*  BSUM   = SM + OFF_BSUM;
    unsigned*  BCOUT  = SM + OFF_BCOUT;
    float*     BCARRY = (float*)(SM + OFF_BCARRY);
    _Float16*  ELDS   = (_Float16*)(SM + OFF_E);
    float*     CARRYF = (float*)(SM + OFF_CARRY);

    const int tid  = threadIdx.x;
    const int wave = tid >> 5;
    const int lane = tid & 31;
    const int half = lane >> 4;    // 0: lanes 0-15, 1: lanes 16-31
    const int m    = lane & 15;    // A-side row / C-side column index

    // ======== one-time: build packed-f16 / f32 B fragments from weights ======
    // f16 B-fragment layout (mirror of documented A layout, roles col<->lane):
    //   lanes 0-15 : V0..V3 hold K=0..7 (pairs), V4..V7 hold K=16..23, col=lane
    //   lanes16-31 : V0..V3 hold K=8..15,        V4..V7 hold K=24..31, col=lane-16
    for (int idx = tid; idx < 32*32*8; idx += BLOCK_THREADS) {
        int r  = idx & 7, ln = (idx >> 3) & 31, t = idx >> 8;
        int col = t*16 + (ln & 15);
        int k0  = ((r < 4) ? 2*r : 2*r + 8) + ((ln >> 4) ? 8 : 0);
        BIN0[idx] = pack2h(Win[k0*512 + col], Win[(k0+1)*512 + col]);
    }
    for (int idx = tid; idx < 16*32*8; idx += BLOCK_THREADS) {
        int r = idx & 7, ln = (idx >> 3) & 31, c = idx >> 8;
        int n  = ln & 15;
        int k0 = c*32 + ((r < 4) ? 2*r : 2*r + 8) + ((ln >> 4) ? 8 : 0);
        BSUM[idx]  = pack2h(Wsum[k0*16 + n], Wsum[(k0+1)*16 + n]);
        // col 0,1: carry weights; col 2: ones (row-sum -> softmax Z); rest 0
        BCOUT[idx] = (n < 2) ? pack2h(Wcout[k0*2 + n], Wcout[(k0+1)*2 + n])
                   : (n == 2) ? pack2h(1.0f, 1.0f) : 0u;
    }
    // f32 16x16x4 B fragment: V0={K0|K2}, V1={K1|K3}; K2,K3 are zero padding,
    // so lanes 16-31 get pre-baked zeros -> unconditional ds_load_b64 in loop.
    for (int idx = tid; idx < 32*32; idx += BLOCK_THREADS) {
        int t = idx >> 5, ln = idx & 31;
        int col = t*16 + (ln & 15);
        BCARRY[idx*2 + 0] = (ln < 16) ? Win[32*512 + col] : 0.0f;
        BCARRY[idx*2 + 1] = (ln < 16) ? Win[33*512 + col] : 0.0f;
    }
    if (lane < 16) {                                   // carry0 = [1, 0]
        CARRYF[(wave*16 + lane)*2 + 0] = 1.0f;
        CARRYF[(wave*16 + lane)*2 + 1] = 0.0f;
    }
    __syncthreads();

    const long sampleBase = (long)blockIdx.x * (WAVES_PER_BLOCK*16) + wave*16;
    _Float16* ew = ELDS + wave * 16 * 512;

#pragma unroll 1
    for (int j = 0; j < NSTEP; ++j) {
        // ---- A1: [16 x 32] f16, K0-15 = a nibble one-hot, K16-31 = b one-hot
        const float* aRow = a_oh + ((sampleBase + m)*NSTEP + j)*16 + half*8;
        const float* bRow = b_oh + ((sampleBase + m)*NSTEP + j)*16 + half*8;
        float4 a0 = ((const float4*)aRow)[0], a1 = ((const float4*)aRow)[1];
        float4 b0 = ((const float4*)bRow)[0], b1 = ((const float4*)bRow)[1];
        if (j + 1 < NSTEP) {                       // stream-ahead hint
            __builtin_prefetch(aRow + 16, 0, 0);
            __builtin_prefetch(bRow + 16, 0, 0);
        }
        v16h A1;
        A1[0]=(_Float16)a0.x;  A1[1]=(_Float16)a0.y;  A1[2]=(_Float16)a0.z;  A1[3]=(_Float16)a0.w;
        A1[4]=(_Float16)a1.x;  A1[5]=(_Float16)a1.y;  A1[6]=(_Float16)a1.z;  A1[7]=(_Float16)a1.w;
        A1[8]=(_Float16)b0.x;  A1[9]=(_Float16)b0.y;  A1[10]=(_Float16)b0.z; A1[11]=(_Float16)b0.w;
        A1[12]=(_Float16)b1.x; A1[13]=(_Float16)b1.y; A1[14]=(_Float16)b1.z; A1[15]=(_Float16)b1.w;

        // ---- A2: carry [16 x 4] f32.  A layout: lanes0-15 V0=K0,V1=K1;
        //      lanes16-31 V0=K2,V1=K3 (zero).
        float c0 = CARRYF[(wave*16 + m)*2 + 0];
        float c1 = CARRYF[(wave*16 + m)*2 + 1];
        v2f A2 = { half ? 0.0f : c0, half ? 0.0f : c1 };

        // ---- GEMM 1: logits [16 x 512] as 32 column tiles, exp + transpose --
        // Single per-tile base pointer; r*512 displacements become DS
        // immediate offsets (max 7168 < 64K) -> 1 address add per tile.
        _Float16* ebase0 = ew + (half*8)*512 + m;
#pragma unroll 2
        for (int t = 0; t < 32; ++t) {
            Frag16 B0; B0.u = *(const v8u*)&BIN0[(t*32 + lane)*8];
            v2f Bc = *(const v2f*)&BCARRY[(t*32 + lane)*2];  // zeros pre-baked

            v8f acc = {};
            acc = __builtin_amdgcn_wmma_f32_16x16x32_f16(false, A1, false, B0.h,
                                                         (short)0, acc, false, false);
            acc = __builtin_amdgcn_wmma_f32_16x16x4_f32(false, A2, false, Bc,
                                                        (short)0, acc, false, false);
            // softmax numerator (max logit == 3 exactly for one-hot inputs)
            _Float16* ebase = ebase0 + t*16;
#pragma unroll
            for (int r = 0; r < 8; ++r) {
                float e = fast_exp2(fmaf(acc[r], SCALE_L2E, BIAS_L2E));
                // C layout: row = r + 8*half, col = t*16 + m  -> LDS transpose
                ebase[r*512] = (_Float16)e;
            }
        }
        // wave-local ordering of E-buffer stores vs. reloads (no s_barrier
        // needed: E and CARRY regions are wave-private)
        asm volatile("s_wait_dscnt 0x0" ::: "memory");

        // ---- GEMM 2: e[16x512] @ [Wcout | ones | 0...] and @ Wsum -----------
        v8f accS = {}, accC = {};
        const _Float16* rowbase = ew + m*512 + half*8;
#pragma unroll 2
        for (int c = 0; c < 16; ++c) {
            const _Float16* rowp = rowbase + c*32;
            Frag16 Af;
            Af.q[0] = *(const uint4*)(rowp);
            Af.q[1] = *(const uint4*)(rowp + 16);
            Frag16 Bs; Bs.u = *(const v8u*)&BSUM[(c*32 + lane)*8];
            Frag16 Bc; Bc.u = *(const v8u*)&BCOUT[(c*32 + lane)*8];
            accS = __builtin_amdgcn_wmma_f32_16x16x32_f16(false, Af.h, false, Bs.h,
                                                          (short)0, accS, false, false);
            accC = __builtin_amdgcn_wmma_f32_16x16x32_f16(false, Af.h, false, Bc.h,
                                                          (short)0, accC, false, false);
        }

        // ---- normalize by Z (accC col 2) + emit sum nibble, update carry ----
#pragma unroll
        for (int r = 0; r < 8; ++r) {
            float z = __shfl(accC[r], 2, 16);        // broadcast Z[row] per half
            float zi = 1.0f / z;
            float o = accS[r] * zi;                  // row r+8*half, col m
            out[((sampleBase + r + 8*half)*NSTEP + j)*16 + m] = o;
            float cv = accC[r] * zi;                 // col m: 0 -> c0, 1 -> c1
            if (m < 2) CARRYF[(wave*16 + r + 8*half)*2 + m] = cv;
        }
        asm volatile("s_wait_dscnt 0x0" ::: "memory");   // carry RAW ordering
    }
}

extern "C" void kernel_launch(void* const* d_in, const int* in_sizes, int n_in,
                              void* d_out, int out_size, void* d_ws, size_t ws_size,
                              hipStream_t stream) {
    const float* a_oh  = (const float*)d_in[0];
    const float* b_oh  = (const float*)d_in[1];
    const float* Win   = (const float*)d_in[2];
    const float* Wsum  = (const float*)d_in[3];
    const float* Wcout = (const float*)d_in[4];
    float* out = (float*)d_out;

    const int n = in_sizes[0] / (NSTEP * 16);            // N samples
    const int samplesPerBlock = WAVES_PER_BLOCK * 16;
    const int blocks = n / samplesPerBlock;              // N divides evenly

    static bool attr_set = false;  // idempotent config, not a stream op
    if (!attr_set) {
        hipFuncSetAttribute((const void*)nibble32_wmma_kernel,
                            hipFuncAttributeMaxDynamicSharedMemorySize, SMEM_BYTES);
        attr_set = true;
    }
    nibble32_wmma_kernel<<<dim3(blocks), dim3(BLOCK_THREADS), SMEM_BYTES, stream>>>(
        a_oh, b_oh, Win, Wsum, Wcout, out);
}
